// SpMiddlePillarEncoder_8942121910492
// MI455X (gfx1250) — compile-verified, matrix operand order checked by reading, not run on hardware
//
#include <hip/hip_runtime.h>

// ---------------------------------------------------------------------------
// Types for gfx1250 WMMA (wave32): v_wmma_f32_16x16x32_bf16
// ---------------------------------------------------------------------------
typedef __bf16 bf16_t;
typedef bf16_t v16bf __attribute__((ext_vector_type(16)));
typedef float  v8f   __attribute__((ext_vector_type(8)));
typedef unsigned int u32x4 __attribute__((ext_vector_type(4)));

// ---------------------------------------------------------------------------
// MAC loop over 9 taps x (CI/32) K-chunks. SAFE=true -> interior tile, no
// bounds handling whatsoever; SAFE=false -> border tile, OOB taps read a
// zeroed page via a 64-bit address select.
// Fragment layout (gfx1250 16-bit striping): lane half h owns elements
//   [8h .. 8h+7] and [16+8h .. 23+8h]  ->  exactly two b128 loads per fragment.
// ---------------------------------------------------------------------------
template<int STRIDE, int CI, bool SAFE>
__device__ __forceinline__ void conv_mac(const bf16_t* __restrict__ in,
                                         const bf16_t* __restrict__ wrow0,
                                         const bf16_t* __restrict__ wrow1,
                                         const bf16_t* __restrict__ zpg,
                                         int bb, int yo, int xo, int Hi, int Wi,
                                         int hf, v8f& acc0, v8f& acc1)
{
    #pragma unroll
    for (int tap = 0; tap < 9; ++tap) {
        const int ky = tap / 3 - 1;
        const int kx = tap % 3 - 1;
        const int yi = yo * STRIDE + ky;
        const int xi = xo * STRIDE + kx;
        const bf16_t* aBase;
        if (SAFE) {
            aBase = in + ((size_t)(bb * Hi + yi) * Wi + xi) * CI;
        } else {
            const bool ok = ((unsigned)yi < (unsigned)Hi) && ((unsigned)xi < (unsigned)Wi);
            aBase = ok ? (in + ((size_t)(bb * Hi + yi) * Wi + xi) * CI) : zpg;
        }
        const bf16_t* b0 = wrow0 + tap * CI;
        const bf16_t* b1 = wrow1 + tap * CI;
        if (tap < 8) __builtin_prefetch(wrow0 + (tap + 1) * CI);  // global_prefetch_b8

        #pragma unroll
        for (int c0 = 0; c0 < CI; c0 += 32) {
            v16bf av, bv0, bv1;
            ((u32x4*)&av)[0]  = *(const u32x4*)(aBase + c0 + 8 * hf);
            ((u32x4*)&av)[1]  = *(const u32x4*)(aBase + c0 + 16 + 8 * hf);
            ((u32x4*)&bv0)[0] = *(const u32x4*)(b0 + c0 + 8 * hf);
            ((u32x4*)&bv0)[1] = *(const u32x4*)(b0 + c0 + 16 + 8 * hf);
            ((u32x4*)&bv1)[0] = *(const u32x4*)(b1 + c0 + 8 * hf);
            ((u32x4*)&bv1)[1] = *(const u32x4*)(b1 + c0 + 16 + 8 * hf);
            acc0 = __builtin_amdgcn_wmma_f32_16x16x32_bf16(
                false, av, false, bv0, (short)0, acc0, false, false);
            acc1 = __builtin_amdgcn_wmma_f32_16x16x32_bf16(
                false, av, false, bv1, (short)0, acc1, false, false);
        }
    }
}

// ---------------------------------------------------------------------------
// Implicit-GEMM 3x3 conv. One wave computes a 16(spatial) x 32(Co) tile:
//   A (16x32 bf16): 16 consecutive output x positions, K = 32 input channels
//                   of one 3x3 tap (CI is a multiple of 32, so every K-chunk
//                   stays inside a single tap).
//   B (32x16 bf16) x2: weights pre-transposed to [Co][9][CI]; one A fragment
//                   feeds two back-to-back WMMAs (Co and Co+16).
// Epilogue fuses BN affine + ReLU + active-site mask + residual:
//   MODE 0: y = bn(conv) * m
//   MODE 1: y = relu(bn(conv)) * m
//   MODE 2: y = relu(bn(conv) * m + resid)
// ---------------------------------------------------------------------------
template<int MODE, int STRIDE, int CI>
__global__ __launch_bounds__(128)
void conv3x3_wmma(const bf16_t* __restrict__ in,     // [B,Hi,Wi,CI]
                  const bf16_t* __restrict__ wT,     // [Co,9,CI]
                  const float*  __restrict__ scale,  // [Co]
                  const float*  __restrict__ shift,  // [Co]
                  const float*  __restrict__ mask,   // [B,Ho,Wo]
                  const bf16_t* __restrict__ resid,  // [B,Ho,Wo,Co] (MODE 2)
                  const bf16_t* __restrict__ zpg,    // 1KiB zero page
                  bf16_t*       __restrict__ out,    // [B,Ho,Wo,Co]
                  int Bn, int Hi, int Wi, int Ho, int Wo, int Co)
{
    const int lane   = threadIdx.x;
    const int tilesX = Wo >> 4;                       // Wo is a multiple of 16
    const int wave   = blockIdx.x * blockDim.y + threadIdx.y;
    if (wave >= Bn * Ho * tilesX) return;             // wave-uniform: EXEC stays full
    const int tx  = wave % tilesX;
    const int rem = wave / tilesX;
    const int yo  = rem % Ho;
    const int bb  = rem / Ho;
    const int x0  = tx << 4;
    const int coBase = blockIdx.y << 5;               // 32 output channels per wave

    const int nIdx = lane & 15;                       // A-row M / B-col N / D-col N
    const int hf   = lane >> 4;                       // lane half -> K sub-range
    const int xo   = x0 + nIdx;                       // this lane's spatial position
    const int K    = 9 * CI;

    const bf16_t* wrow0 = wT + (size_t)(coBase + nIdx) * K;
    const bf16_t* wrow1 = wrow0 + (size_t)16 * K;

    v8f acc0 = {}, acc1 = {};
    // Wave-uniform interior test: all 9 taps in bounds for all 16 lanes' x.
    const bool interior = (yo * STRIDE - 1 >= 0) && (yo * STRIDE + 1 < Hi) &&
                          (x0 * STRIDE - 1 >= 0) && ((x0 + 15) * STRIDE + 1 < Wi);
    if (interior)
        conv_mac<STRIDE, CI, true >(in, wrow0, wrow1, zpg, bb, yo, xo, Hi, Wi, hf, acc0, acc1);
    else
        conv_mac<STRIDE, CI, false>(in, wrow0, wrow1, zpg, bb, yo, xo, Hi, Wi, hf, acc0, acc1);

    // Epilogue: lane owns channels coBase+nIdx and coBase+nIdx+16.
    const int   co0 = coBase + nIdx;
    const int   co1 = co0 + 16;
    const float s0 = scale[co0], sh0 = shift[co0];
    const float s1 = scale[co1], sh1 = shift[co1];
    const size_t rowBase = (size_t)(bb * Ho + yo) * Wo;
    #pragma unroll
    for (int r = 0; r < 8; ++r) {
        const int    m = r + 8 * hf;                  // D row -> spatial offset
        const size_t p = rowBase + (x0 + m);
        const float mk = mask[p];
        float v0 = acc0[r] * s0 + sh0;
        float v1 = acc1[r] * s1 + sh1;
        float y0, y1;
        if (MODE == 0)      { y0 = v0 * mk;             y1 = v1 * mk; }
        else if (MODE == 1) { y0 = fmaxf(v0, 0.f) * mk; y1 = fmaxf(v1, 0.f) * mk; }
        else {
            y0 = fmaxf(v0 * mk + (float)resid[p * Co + co0], 0.f);
            y1 = fmaxf(v1 * mk + (float)resid[p * Co + co1], 0.f);
        }
        out[p * Co + co0] = (bf16_t)y0;
        out[p * Co + co1] = (bf16_t)y1;
    }
}

// ---------------------------------------------------------------------------
// Layout / prep kernels
// ---------------------------------------------------------------------------
__global__ void zero_fill(float* __restrict__ p, int n)
{
    int i = blockIdx.x * blockDim.x + threadIdx.x;
    if (i < n) p[i] = 0.f;
}

__global__ void prep_input(const float* __restrict__ x, const int* __restrict__ occ,
                           bf16_t* __restrict__ nhwc, float* __restrict__ m,
                           int Bn, int C, int H, int W)
{
    size_t i = (size_t)blockIdx.x * blockDim.x + threadIdx.x;
    size_t N = (size_t)Bn * H * W * C;
    if (i >= N) return;
    int c = (int)(i % C);
    size_t p = i / C;                       // (b*H+y)*W + x
    int xx = (int)(p % W);
    size_t q = p / W;
    int y = (int)(q % H);
    int b = (int)(q / H);
    float mk = (occ[p] == 0) ? 1.f : 0.f;
    float v  = x[((size_t)(b * C + c) * H + y) * W + xx];
    nhwc[i] = (bf16_t)(v * mk);
    if (c == 0) m[p] = mk;
}

__global__ void prep_weight(const float* __restrict__ w,   // HWIO [3,3,Ci,Co] f32
                            bf16_t* __restrict__ wT,       // [Co,9,Ci] bf16
                            int Ci, int Co)
{
    int i = blockIdx.x * blockDim.x + threadIdx.x;
    int N = 9 * Ci * Co;
    if (i >= N) return;
    int co  = i / (9 * Ci);
    int r   = i % (9 * Ci);                 // tap*Ci + ci
    wT[i] = (bf16_t)w[(size_t)r * Co + co]; // w[(tap*Ci+ci)*Co + co]
}

__global__ void mask_down(const float* __restrict__ mi, float* __restrict__ mo,
                          int Bn, int Hi, int Wi, int Ho, int Wo)
{
    int i = blockIdx.x * blockDim.x + threadIdx.x;
    if (i >= Bn * Ho * Wo) return;
    int xo = i % Wo; int t = i / Wo; int yo = t % Ho; int b = t / Ho;
    float any = 0.f;
    #pragma unroll
    for (int dy = -1; dy <= 1; ++dy)
        #pragma unroll
        for (int dx = -1; dx <= 1; ++dx) {
            int yi = 2 * yo + dy, xi = 2 * xo + dx;
            if ((unsigned)yi < (unsigned)Hi && (unsigned)xi < (unsigned)Wi)
                any += mi[((size_t)(b * Hi + yi)) * Wi + xi];
        }
    mo[i] = (any > 0.f) ? 1.f : 0.f;
}

__global__ void to_nchw(const bf16_t* __restrict__ nhwc, float* __restrict__ out,
                        int Bn, int C, int H, int W)
{
    size_t i = (size_t)blockIdx.x * blockDim.x + threadIdx.x;
    if (i >= (size_t)Bn * C * H * W) return;
    int xx = (int)(i % W); size_t t = i / W;
    int y  = (int)(t % H); t /= H;
    int c  = (int)(t % C);
    int b  = (int)(t / C);
    out[i] = (float)nhwc[((size_t)(b * H + y) * W + xx) * C + c];
}

// ---------------------------------------------------------------------------
// Host orchestration
// ---------------------------------------------------------------------------
template<int CI>
static void launch_conv_ci(int mode, int stride,
                           const bf16_t* in, const bf16_t* wT,
                           const float* s, const float* b, const float* m,
                           const bf16_t* resid, const bf16_t* zpg, bf16_t* out,
                           int Bn, int Hi, int Wi, int Ho, int Wo, int Co,
                           hipStream_t st)
{
    long waves = (long)Bn * Ho * (Wo >> 4);
    dim3 blk(32, 4), grd((unsigned)((waves + 3) / 4), (unsigned)(Co >> 5));
    if (mode == 0)
        conv3x3_wmma<0,1,CI><<<grd, blk, 0, st>>>(in, wT, s, b, m, resid, zpg, out, Bn, Hi, Wi, Ho, Wo, Co);
    else if (mode == 1 && stride == 1)
        conv3x3_wmma<1,1,CI><<<grd, blk, 0, st>>>(in, wT, s, b, m, resid, zpg, out, Bn, Hi, Wi, Ho, Wo, Co);
    else if (mode == 2)
        conv3x3_wmma<2,1,CI><<<grd, blk, 0, st>>>(in, wT, s, b, m, resid, zpg, out, Bn, Hi, Wi, Ho, Wo, Co);
    else
        conv3x3_wmma<1,2,CI><<<grd, blk, 0, st>>>(in, wT, s, b, m, resid, zpg, out, Bn, Hi, Wi, Ho, Wo, Co);
}

static void launch_conv(int Ci, int mode, int stride,
                        const bf16_t* in, const bf16_t* wT,
                        const float* s, const float* b, const float* m,
                        const bf16_t* resid, const bf16_t* zpg, bf16_t* out,
                        int Bn, int Hi, int Wi, int Ho, int Wo, int Co,
                        hipStream_t st)
{
    switch (Ci) {
    case 32:  launch_conv_ci<32> (mode, stride, in, wT, s, b, m, resid, zpg, out, Bn, Hi, Wi, Ho, Wo, Co, st); break;
    case 64:  launch_conv_ci<64> (mode, stride, in, wT, s, b, m, resid, zpg, out, Bn, Hi, Wi, Ho, Wo, Co, st); break;
    case 128: launch_conv_ci<128>(mode, stride, in, wT, s, b, m, resid, zpg, out, Bn, Hi, Wi, Ho, Wo, Co, st); break;
    default:  launch_conv_ci<256>(mode, stride, in, wT, s, b, m, resid, zpg, out, Bn, Hi, Wi, Ho, Wo, Co, st); break;
    }
}

extern "C" void kernel_launch(void* const* d_in, const int* in_sizes, int n_in,
                              void* d_out, int out_size, void* d_ws, size_t ws_size,
                              hipStream_t stream)
{
    // ---- input layout per JAX pytree (sorted-key) flattening ----
    // 0:x 1:occ | blk1:2-7 blk2a:8-13 blk2b:14-19 blk3a:20-25 blk3b:26-31
    // blk4a:32-37 blk4b:38-43 (each: b1,b2,s1,s2,w1,w2)
    // down2:44-46 down3:47-49 down4:50-52 (b,s,w)
    // v:53-61 (b0,b1,b2,s0,s1,s2,w0,w1,w2)
    auto F = [&](int i) { return (const float*)d_in[i]; };
    const float* X   = (const float*)d_in[0];
    const int*   OCC = (const int*)d_in[1];

    const size_t MiB = 1024u * 1024u;
    if (ws_size < 112 * MiB) return;  // deterministic guard
    char* ws = (char*)d_ws;
    bf16_t* fA  = (bf16_t*)(ws +   0 * MiB);
    bf16_t* fB  = (bf16_t*)(ws +  32 * MiB);
    bf16_t* fC  = (bf16_t*)(ws +  64 * MiB);
    float*  m1  = (float*) (ws +  96 * MiB);   // 2 MiB
    bf16_t* zpg = (bf16_t*)(ws +  98 * MiB);   // 1 KiB zero page for OOB taps
    float*  m2  = (float*) (ws +  99 * MiB);   // 512 KiB
    float*  m3  = (float*) (ws + 100 * MiB);   // 128 KiB
    float*  m4  = (float*) (ws + 101 * MiB);   // 32 KiB
    char*   wp  = ws + 102 * MiB;              // bf16 transposed weights (~7 MB)

    zero_fill<<<1, 256, 0, stream>>>((float*)zpg, 256);

    auto prepW = [&](const float* w, int Ci, int Co) -> bf16_t* {
        bf16_t* dst = (bf16_t*)wp;
        wp += (size_t)9 * Ci * Co * sizeof(bf16_t);
        int n = 9 * Ci * Co;
        prep_weight<<<(n + 255) / 256, 256, 0, stream>>>(w, dst, Ci, Co);
        return dst;
    };

    // ---- weight transforms (resident in L2 thereafter) ----
    bf16_t* w_v0   = prepW(F(59), 32, 32);
    bf16_t* w_v1   = prepW(F(60), 32, 32);
    bf16_t* w_v2   = prepW(F(61), 32, 32);
    bf16_t* w_b11  = prepW(F(6),  32, 32);
    bf16_t* w_b12  = prepW(F(7),  32, 32);
    bf16_t* w_d2   = prepW(F(46), 32, 64);
    bf16_t* w_b2a1 = prepW(F(12), 64, 64);
    bf16_t* w_b2a2 = prepW(F(13), 64, 64);
    bf16_t* w_b2b1 = prepW(F(18), 64, 64);
    bf16_t* w_b2b2 = prepW(F(19), 64, 64);
    bf16_t* w_d3   = prepW(F(49), 64, 128);
    bf16_t* w_b3a1 = prepW(F(24), 128, 128);
    bf16_t* w_b3a2 = prepW(F(25), 128, 128);
    bf16_t* w_b3b1 = prepW(F(30), 128, 128);
    bf16_t* w_b3b2 = prepW(F(31), 128, 128);
    bf16_t* w_d4   = prepW(F(52), 128, 256);
    bf16_t* w_b4a1 = prepW(F(36), 256, 256);
    bf16_t* w_b4a2 = prepW(F(37), 256, 256);
    bf16_t* w_b4b1 = prepW(F(42), 256, 256);
    bf16_t* w_b4b2 = prepW(F(43), 256, 256);

    // ---- input: NCHW f32 -> masked NHWC bf16, mask m1 ----
    {
        size_t n = (size_t)2 * 512 * 512 * 32;
        prep_input<<<(unsigned)((n + 255) / 256), 256, 0, stream>>>(X, OCC, fA, m1, 2, 32, 512, 512);
    }

    // ---- stage 1 (512x512, C=32, SubM) ----
    launch_conv(32, 0,1, fA, w_v0,  F(56), F(53), m1, nullptr, zpg, fB, 2,512,512, 512,512,32, stream);
    launch_conv(32, 1,1, fB, w_v1,  F(57), F(54), m1, nullptr, zpg, fC, 2,512,512, 512,512,32, stream);
    launch_conv(32, 2,1, fC, w_v2,  F(58), F(55), m1, fB,      zpg, fA, 2,512,512, 512,512,32, stream);
    launch_conv(32, 1,1, fA, w_b11, F(4),  F(2),  m1, nullptr, zpg, fB, 2,512,512, 512,512,32, stream);
    launch_conv(32, 2,1, fB, w_b12, F(5),  F(3),  m1, fA,      zpg, fC, 2,512,512, 512,512,32, stream);

    // ---- stage 2 (256x256, C=64) ----
    { int n = 2*256*256; mask_down<<<(n+255)/256, 256, 0, stream>>>(m1, m2, 2, 512,512, 256,256); }
    launch_conv(32, 1,2, fC, w_d2,   F(45), F(44), m2, nullptr, zpg, fA, 2,512,512, 256,256,64, stream);
    launch_conv(64, 1,1, fA, w_b2a1, F(10), F(8),  m2, nullptr, zpg, fB, 2,256,256, 256,256,64, stream);
    launch_conv(64, 2,1, fB, w_b2a2, F(11), F(9),  m2, fA,      zpg, fC, 2,256,256, 256,256,64, stream);
    launch_conv(64, 1,1, fC, w_b2b1, F(16), F(14), m2, nullptr, zpg, fA, 2,256,256, 256,256,64, stream);
    launch_conv(64, 2,1, fA, w_b2b2, F(17), F(15), m2, fC,      zpg, fB, 2,256,256, 256,256,64, stream);

    // ---- stage 3 (128x128, C=128) ----
    { int n = 2*128*128; mask_down<<<(n+255)/256, 256, 0, stream>>>(m2, m3, 2, 256,256, 128,128); }
    launch_conv(64,  1,2, fB, w_d3,   F(48), F(47), m3, nullptr, zpg, fC, 2,256,256, 128,128,128, stream);
    launch_conv(128, 1,1, fC, w_b3a1, F(22), F(20), m3, nullptr, zpg, fA, 2,128,128, 128,128,128, stream);
    launch_conv(128, 2,1, fA, w_b3a2, F(23), F(21), m3, fC,      zpg, fB, 2,128,128, 128,128,128, stream);
    launch_conv(128, 1,1, fB, w_b3b1, F(28), F(26), m3, nullptr, zpg, fC, 2,128,128, 128,128,128, stream);
    launch_conv(128, 2,1, fC, w_b3b2, F(29), F(27), m3, fB,      zpg, fA, 2,128,128, 128,128,128, stream);

    // ---- stage 4 (64x64, C=256) ----
    { int n = 2*64*64; mask_down<<<(n+255)/256, 256, 0, stream>>>(m3, m4, 2, 128,128, 64,64); }
    launch_conv(128, 1,2, fA, w_d4,   F(51), F(50), m4, nullptr, zpg, fB, 2,128,128, 64,64,256, stream);
    launch_conv(256, 1,1, fB, w_b4a1, F(34), F(32), m4, nullptr, zpg, fC, 2,64,64,   64,64,256, stream);
    launch_conv(256, 2,1, fC, w_b4a2, F(35), F(33), m4, fB,      zpg, fA, 2,64,64,   64,64,256, stream);
    launch_conv(256, 1,1, fA, w_b4b1, F(40), F(38), m4, nullptr, zpg, fC, 2,64,64,   64,64,256, stream);
    launch_conv(256, 2,1, fC, w_b4b2, F(41), F(39), m4, fA,      zpg, fB, 2,64,64,   64,64,256, stream);

    // ---- output: NHWC bf16 -> NCHW f32 ----
    {
        size_t n = (size_t)2 * 256 * 64 * 64;
        to_nchw<<<(unsigned)((n + 255) / 256), 256, 0, stream>>>(fB, (float*)d_out, 2, 256, 64, 64);
    }
}
